// DualSTN_73315091743300
// MI455X (gfx1250) — matrix operand order repaired and checked by reading, compiled, not run on hardware
//
#include <hip/hip_runtime.h>
#include <hip/hip_bf16.h>
#include <math.h>

// ---------------------------------------------------------------------------
// DualSTN forward for MI455X (gfx1250, wave32, WGP).
// All matmuls run through V_WMMA_F32_16X16X32_BF16 (bf16 A/B, f32 accum).
// GEMM: 128-thread block (4 waves) -> 64x64 macro-tile; A/B panels staged
// through LDS in WMMA-fragment-friendly bf16 layout; each wave holds one
// 16-row subtile x four 16-col subtiles (4 accumulators, 4 wmma / K-step).
// ---------------------------------------------------------------------------

typedef __attribute__((ext_vector_type(16))) __bf16 v16bf;
typedef __attribute__((ext_vector_type(8)))  float v8f;
typedef __attribute__((ext_vector_type(8)))  unsigned int v8ui;

#define AL_C 0.05f
#define BE_C 0.95f
#define GA_C 0.95f
#define TAU_C 2.0f
#define LDSS 34   // LDS row stride in halves (32 + 2 pad, even for b32 reads)

static __device__ __forceinline__ unsigned short f2bf(float f) {
  unsigned u = __builtin_bit_cast(unsigned, f);
  u += 0x7FFFu + ((u >> 16) & 1u);            // round-to-nearest-even
  return (unsigned short)(u >> 16);
}

// General GEMM: Out[b] = act(alpha * opA(A[b]) @ opB(B[b]) + bias) [+= if accumulate]
//   opA(m,k) = transA ? A[k*lda+m] : A[m*lda+k]
//   opB(k,n) = transB ? B[n*ldb+k] : B[k*ldb+n]
//   Out index = transOut ? n*ldo+m : m*ldo+n  (plus batch strides sA/sB/sO)
// act: 0=none 1=relu 2=sigmoid 3=tanh
__global__ void __launch_bounds__(128) wmma_gemm_kernel(
    const float* __restrict__ A, const float* __restrict__ Bm,
    const float* __restrict__ bias, float* __restrict__ Out,
    int M, int Nn, int K,
    long sA, long sB, long sO,
    int lda, int ldb, int ldo,
    int transA, int transB, int transOut,
    float alpha, int accumulate, int act)
{
  __shared__ __align__(16) unsigned short lA[64 * LDSS];  // (m, k) bf16
  __shared__ __align__(16) unsigned short lB[64 * LDSS];  // (n, k) bf16
  const int tid  = threadIdx.x;
  const int wid  = tid >> 5;
  const int lane = tid & 31;
  const int half = lane >> 4;
  const int lm   = lane & 15;
  const int b    = blockIdx.z;
  const int mBlk = blockIdx.x * 64;
  const int nBlk = blockIdx.y * 64;
  const float* Ab = A + (long)b * sA;
  const float* Bb = Bm + (long)b * sB;
  float* Ob = Out + (long)b * sO;

  v8f acc0 = {0.f,0.f,0.f,0.f,0.f,0.f,0.f,0.f};
  v8f acc1 = acc0, acc2 = acc0, acc3 = acc0;

  // WMMA bf16 lane layouts (ISA 7.12.2): A 16x32 per-lane K pattern, B 32x16.
  const int baseKA = half ? 16 : 0;
  const int baseKB = half ? 16 : 0;
  const int mRow   = wid * 16 + lm;   // A-fragment row within macro-tile

  for (int k0 = 0; k0 < K; k0 += 32) {
    const bool fullK = (k0 + 32 <= K);
    // ---- stage A panel (64 m x 32 k) -> lA[m][k] ----
    if (transA) {
      if ((mBlk + 64 <= M) && fullK) {
        for (int idx = tid; idx < 64 * 32; idx += 128) {
          int kk = idx >> 6, mm = idx & 63;   // coalesce along m
          lA[mm * LDSS + kk] = f2bf(Ab[(long)(k0 + kk) * lda + (mBlk + mm)]);
        }
      } else {
        for (int idx = tid; idx < 64 * 32; idx += 128) {
          int kk = idx >> 6, mm = idx & 63;
          int kc = k0 + kk, mc = mBlk + mm;
          bool in = (kc < K) && (mc < M);
          float v = Ab[(long)min(kc, K - 1) * lda + min(mc, M - 1)];
          lA[mm * LDSS + kk] = f2bf(in ? v : 0.f);
        }
      }
    } else {
      if ((mBlk + 64 <= M) && fullK) {
        for (int idx = tid; idx < 64 * 32; idx += 128) {
          int kk = idx & 31, mm = idx >> 5;   // coalesce along k
          lA[mm * LDSS + kk] = f2bf(Ab[(long)(mBlk + mm) * lda + (k0 + kk)]);
        }
      } else {
        for (int idx = tid; idx < 64 * 32; idx += 128) {
          int kk = idx & 31, mm = idx >> 5;
          int kc = k0 + kk, mc = mBlk + mm;
          bool in = (kc < K) && (mc < M);
          float v = Ab[(long)min(mc, M - 1) * lda + min(kc, K - 1)];
          lA[mm * LDSS + kk] = f2bf(in ? v : 0.f);
        }
      }
    }
    // ---- stage B panel (32 k x 64 n) -> lB[n][k] (transposed in LDS) ----
    if (transB) {
      if ((nBlk + 64 <= Nn) && fullK) {
        for (int idx = tid; idx < 64 * 32; idx += 128) {
          int kk = idx & 31, nn = idx >> 5;   // coalesce along k
          lB[nn * LDSS + kk] = f2bf(Bb[(long)(nBlk + nn) * ldb + (k0 + kk)]);
        }
      } else {
        for (int idx = tid; idx < 64 * 32; idx += 128) {
          int kk = idx & 31, nn = idx >> 5;
          int kc = k0 + kk, nc = nBlk + nn;
          bool in = (kc < K) && (nc < Nn);
          float v = Bb[(long)min(nc, Nn - 1) * ldb + min(kc, K - 1)];
          lB[nn * LDSS + kk] = f2bf(in ? v : 0.f);
        }
      }
    } else {
      if ((nBlk + 64 <= Nn) && fullK) {
        for (int idx = tid; idx < 64 * 32; idx += 128) {
          int nn = idx & 63, kk = idx >> 6;   // coalesce along n
          lB[nn * LDSS + kk] = f2bf(Bb[(long)(k0 + kk) * ldb + (nBlk + nn)]);
        }
      } else {
        for (int idx = tid; idx < 64 * 32; idx += 128) {
          int nn = idx & 63, kk = idx >> 6;
          int kc = k0 + kk, nc = nBlk + nn;
          bool in = (kc < K) && (nc < Nn);
          float v = Bb[(long)min(kc, K - 1) * ldb + min(nc, Nn - 1)];
          lB[nn * LDSS + kk] = f2bf(in ? v : 0.f);
        }
      }
    }
    __syncthreads();

    // ---- build fragments from LDS (aligned 32-bit reads) and WMMA ----
    v8ui au;
#pragma unroll
    for (int p = 0; p < 8; ++p) {
      int kk = baseKA + 2 * p;   // k-pairs
      au[p] = *(const unsigned int*)&lA[mRow * LDSS + kk];
    }
    v16bf av = __builtin_bit_cast(v16bf, au);
#pragma unroll
    for (int s = 0; s < 4; ++s) {
      v8ui bu;
#pragma unroll
      for (int p = 0; p < 8; ++p) {
        int kk = baseKB + 2 * p;
        bu[p] = *(const unsigned int*)&lB[(s * 16 + lm) * LDSS + kk];
      }
      v16bf bv = __builtin_bit_cast(v16bf, bu);
      v8f* accp = (s == 0) ? &acc0 : (s == 1) ? &acc1 : (s == 2) ? &acc2 : &acc3;
      *accp = __builtin_amdgcn_wmma_f32_16x16x32_bf16(
          false, av, false, bv, (short)0, *accp, false, false);
    }
    __syncthreads();
  }

  // ---- epilogue: D layout VGPR i -> row m0+i (lanes 0-15) / m0+8+i (16-31) ----
  const int m0 = mBlk + wid * 16;
#pragma unroll
  for (int s = 0; s < 4; ++s) {
    v8f acc = (s == 0) ? acc0 : (s == 1) ? acc1 : (s == 2) ? acc2 : acc3;
    int n = nBlk + s * 16 + lm;
    if (n >= Nn) continue;
    float bv = bias ? bias[n] : 0.f;
#pragma unroll
    for (int i = 0; i < 8; ++i) {
      int m = m0 + i + half * 8;
      if (m < M) {
        float v = alpha * acc[i] + bv;
        if (act == 1) v = fmaxf(v, 0.f);
        else if (act == 2) v = 1.f / (1.f + __expf(-v));
        else if (act == 3) v = tanhf(v);
        long idx = transOut ? ((long)n * ldo + m) : ((long)m * ldo + n);
        if (accumulate) Ob[idx] += v; else Ob[idx] = v;
      }
    }
  }
}

// Out[row, ooff..ooff+C) = a*X + bsc*T1 + g*T2  (X may be batch-strided; T1/T2
// contiguous (rows,C); Out row stride ldo).  T1/T2 nullable.
__global__ void ew3_kernel(const float* __restrict__ X, long xBS, int xRS,
                           const float* __restrict__ T1, const float* __restrict__ T2,
                           float a, float bsc, float g,
                           float* __restrict__ Out, int ldo, int ooff,
                           int C, int Nn, long rows)
{
  long idx = (long)blockIdx.x * blockDim.x + threadIdx.x;
  if (idx >= rows * C) return;
  long row = idx / C; int c = (int)(idx % C);
  long b = row / Nn, n = row % Nn;
  float v = a * X[b * xBS + n * (long)xRS + c];
  if (T1) v += bsc * T1[row * C + c];
  if (T2) v += g * T2[row * C + c];
  Out[row * (long)ldo + ooff + c] = v;
}

// Out[row, ooff..ooff+C) = A * B (both contiguous (rows,C))
__global__ void mul_slice_kernel(const float* __restrict__ Aa, const float* __restrict__ Bb,
                                 float* __restrict__ Out, int ldo, int ooff, int C, long rows)
{
  long idx = (long)blockIdx.x * blockDim.x + threadIdx.x;
  if (idx >= rows * C) return;
  long row = idx / C; int c = (int)(idx % C);
  Out[row * (long)ldo + ooff + c] = Aa[row * C + c] * Bb[row * C + c];
}

// hidden = z*hidden + (1-z)*c
__global__ void gru_update_kernel(float* __restrict__ hidden, const float* __restrict__ z,
                                  const float* __restrict__ c, long n)
{
  long idx = (long)blockIdx.x * blockDim.x + threadIdx.x;
  if (idx >= n) return;
  float zz = z[idx];
  hidden[idx] = zz * hidden[idx] + (1.f - zz) * c[idx];
}

// ns/nt: Out = tanh(TAU * ((cur @ W + b) * G)), cur (rows,2) batch-strided,
// W (2,16), G contiguous (rows,16).
__global__ void gate_tanh_kernel(const float* __restrict__ cur, long curBS,
                                 const float* __restrict__ W, const float* __restrict__ bias,
                                 const float* __restrict__ G, float* __restrict__ Out,
                                 int Nn, long rows)
{
  long idx = (long)blockIdx.x * blockDim.x + threadIdx.x;
  if (idx >= rows * 16) return;
  long row = idx >> 4; int c = (int)(idx & 15);
  long b = row / Nn, n = row % Nn;
  const float* xr = cur + b * curBS + n * 2;
  float v = xr[0] * W[c] + xr[1] * W[16 + c] + bias[c];
  Out[idx] = tanhf(TAU_C * (v * G[idx]));
}

// In-place: a -> relu(tanh(TAU*a)) + eye, then row-normalize.  One wave per row.
__global__ void __launch_bounds__(32) adp_norm_kernel(float* __restrict__ A, int Nn)
{
  int n = blockIdx.x, b = blockIdx.y, lane = threadIdx.x;
  float* rowp = A + ((long)b * Nn + n) * (long)Nn;
  float s = 0.f;
  for (int m = lane; m < Nn; m += 32) {
    float v = tanhf(TAU_C * rowp[m]);
    v = fmaxf(v, 0.f);
    if (m == n) v += 1.f;
    rowp[m] = v;
    s += v;
  }
  for (int off = 16; off > 0; off >>= 1) s += __shfl_xor(s, off, 32);
  float inv = 1.f / s;
  for (int m = lane; m < Nn; m += 32) rowp[m] *= inv;
}

// Linear F=2 projection of X[:, tBase..tBase+tCnt) slice: out (B,tCnt,N,C)
__global__ void qk_lin_kernel(const float* __restrict__ X, const float* __restrict__ W,
                              const float* __restrict__ bias, float* __restrict__ Out,
                              int Bb, int tBase, int tCnt, int Nn, int C, int Tt)
{
  long total = (long)Bb * tCnt * Nn * C;
  long idx = (long)blockIdx.x * blockDim.x + threadIdx.x;
  if (idx >= total) return;
  int c = (int)(idx % C);
  long r = idx / C;
  int n = (int)(r % Nn);
  long r2 = r / Nn;
  int tt = (int)(r2 % tCnt);
  int b = (int)(r2 / tCnt);
  const float* xr = X + (((long)b * Tt + tBase + tt) * Nn + n) * 2;
  Out[idx] = xr[0] * W[c] + xr[1] * W[C + c] + bias[c];
}

// scores[i,j] = sum_d tanh(q[i,d]+k[t,j,d])*tw[d]; softmax over j; Aout = adj + attn.
// One wave per (b,i) row; row buffered in LDS.
__global__ void __launch_bounds__(32) attn_kernel(
    const float* __restrict__ q, const float* __restrict__ kk,
    const float* __restrict__ tw, const float* __restrict__ adj,
    float* __restrict__ Aout, int Nn, int t)
{
  int i = blockIdx.x, b = blockIdx.y, lane = threadIdx.x;
  __shared__ float rowbuf[512];
  const float* qi = q + ((long)b * Nn + i) * 8;
  const float* kb = kk + (((long)b * 4 + t) * (long)Nn) * 8;
  float qv[8], tv[8];
#pragma unroll
  for (int d = 0; d < 8; ++d) { qv[d] = qi[d]; tv[d] = tw[d]; }
  float mx = -1e30f;
  for (int j = lane; j < Nn; j += 32) {
    const float* kj = kb + (long)j * 8;
    float s = 0.f;
#pragma unroll
    for (int d = 0; d < 8; ++d) s += tanhf(qv[d] + kj[d]) * tv[d];
    rowbuf[j] = s;
    mx = fmaxf(mx, s);
  }
  for (int off = 16; off > 0; off >>= 1) mx = fmaxf(mx, __shfl_xor(mx, off, 32));
  float sum = 0.f;
  for (int j = lane; j < Nn; j += 32) {
    float e = __expf(rowbuf[j] - mx);
    rowbuf[j] = e;
    sum += e;
  }
  for (int off = 16; off > 0; off >>= 1) sum += __shfl_xor(sum, off, 32);
  float inv = 1.f / sum;
  float* orow = Aout + ((long)b * Nn + i) * (long)Nn;
  const float* arow = adj + (long)i * Nn;
  for (int j = lane; j < Nn; j += 32) orow[j] = arow[j] + rowbuf[j] * inv;
}

// ---------------------------------------------------------------------------
extern "C" void kernel_launch(void* const* d_in, const int* in_sizes, int n_in,
                              void* d_out, int out_size, void* d_ws, size_t ws_size,
                              hipStream_t stream) {
  (void)in_sizes; (void)n_in; (void)out_size; (void)ws_size;
  const int B = 16, T = 24, N = 500, F = 2, H = 64, D = 8, TGN = 32, HYP = 16, LEN = 24;
  const long rows = (long)B * N;

  const float* X      = (const float*)d_in[0];
  const float* adj    = (const float*)d_in[1];
  const float* wq_w   = (const float*)d_in[7];
  const float* wq_b   = (const float*)d_in[8];
  const float* wk_w   = (const float*)d_in[9];
  const float* wk_b   = (const float*)d_in[10];
  const float* trans_w= (const float*)d_in[11];
  const float* tgn_w  = (const float*)d_in[12];
  const float* tgn_b  = (const float*)d_in[13];
  const float* agg1_w = (const float*)d_in[14];
  const float* agg1_b = (const float*)d_in[15];
  const float* agg2_w = (const float*)d_in[16];
  const float* agg2_b = (const float*)d_in[17];
  const float* src_w  = (const float*)d_in[18];
  const float* src_b  = (const float*)d_in[19];
  const float* tgt_w  = (const float*)d_in[20];
  const float* tgt_b  = (const float*)d_in[21];
  const float* gz_w   = (const float*)d_in[22];
  const float* gz_b   = (const float*)d_in[23];
  const float* gr_w   = (const float*)d_in[24];
  const float* gr_b   = (const float*)d_in[25];
  const float* gc_w   = (const float*)d_in[26];
  const float* gc_b   = (const float*)d_in[27];
  const float* fcs_w  = (const float*)d_in[28];
  const float* fcs_b  = (const float*)d_in[29];
  const float* fcm_w  = (const float*)d_in[30];
  const float* fcm_b  = (const float*)d_in[31];
  float* out = (float*)d_out;

  // ---- workspace carve-up (floats); total ~12.7M floats (~51 MB) ----
  float* w = (float*)d_ws;
  long o = 0;
  float* hidden = w + o; o += (long)B * N * H;       // GRU state
  float* adpbuf = w + o; o += (long)B * N * N;       // adaptive adjacency / A_tgn
  float* cat192 = w + o; o += (long)B * N * 192;     // gcn_single(H) concat
  float* cat198 = w + o; o += (long)B * N * 198;     // gcn_rnn concat
  float* cat6   = w + o; o += (long)B * N * 6;       // TGN concat
  float* cat96  = w + o; o += (long)B * N * 96;      // [tar2, hidden]
  float* t1buf  = w + o; o += (long)B * N * 66;
  float* t2buf  = w + o; o += (long)B * N * 66;
  float* comb   = w + o; o += (long)B * N * 66;      // [cur, hidden] / [cur, r*h]
  float* gsb    = w + o; o += (long)B * N * HYP;
  float* gtb    = w + o; o += (long)B * N * HYP;
  float* nsb    = w + o; o += (long)B * N * HYP;
  float* ntb    = w + o; o += (long)B * N * HYP;
  float* zb     = w + o; o += (long)B * N * H;
  float* rb     = w + o; o += (long)B * N * H;
  float* cb     = w + o; o += (long)B * N * H;
  float* qb     = w + o; o += (long)B * N * D;
  float* kb     = w + o; o += (long)B * 4 * N * D;
  float* tar2   = w + o; o += (long)B * N * TGN;
  float* gatb   = w + o; o += (long)B * N * F;

  auto gemm = [&](const float* A, const float* Bm, const float* bias, float* Out,
                  int M, int Nn, int K, long sA, long sB, long sO,
                  int lda, int ldb, int ldo, int transA, int transB, int transOut,
                  float alpha, int accum, int act, int batch) {
    dim3 g((M + 63) / 64, (Nn + 63) / 64, batch);
    hipLaunchKernelGGL(wmma_gemm_kernel, g, dim3(128), 0, stream,
                       A, Bm, bias, Out, M, Nn, K, sA, sB, sO, lda, ldb, ldo,
                       transA, transB, transOut, alpha, accum, act);
  };
  auto ew3 = [&](const float* Xp, long xBS, int xRS, const float* T1p, const float* T2p,
                 float a, float bsc, float g, float* Outp, int ldo, int ooff, int C) {
    long total = rows * C;
    int blocks = (int)((total + 255) / 256);
    hipLaunchKernelGGL(ew3_kernel, dim3(blocks), dim3(256), 0, stream,
                       Xp, xBS, xRS, T1p, T2p, a, bsc, g, Outp, ldo, ooff, C, N, rows);
  };
  // gconv: Out[b] = A[b]^T @ Hin[b];  A is (v,w) row-major, Hin has row stride ldb
  auto gconv = [&](const float* Amat, long sA, const float* Hm, int ldb, int C, float* Outp) {
    gemm(Amat, Hm, nullptr, Outp, N, C, N, sA, (long)N * ldb, (long)N * C,
         N, ldb, C, 1, 0, 0, 1.f, 0, 0, B);
  };
  // gcn_single: builds concat [x, h1, h2] (row length 3C) and applies W,b
  auto gcn_single = [&](const float* Xin, long xBS, int xRS, int C,
                        const float* Amat, long sA, float* cat,
                        const float* Wt, const float* Wb, int outC, float* Outp,
                        int act, int accum) {
    int catLd = 3 * C;
    ew3(Xin, xBS, xRS, nullptr, nullptr, 1.f, 0.f, 0.f, cat, catLd, 0, C);
    gconv(Amat, sA, cat, catLd, C, t1buf);
    ew3(Xin, xBS, xRS, t1buf, nullptr, AL_C, GA_C, 0.f, cat, catLd, C, C);
    gconv(Amat, sA, cat + C, catLd, C, t1buf);
    ew3(Xin, xBS, xRS, t1buf, nullptr, AL_C, GA_C, 0.f, cat, catLd, 2 * C, C);
    gemm(cat, Wt, Wb, Outp, (int)rows, outC, catLd, 0, 0, 0,
         catLd, outC, outC, 0, 0, 0, 1.f, accum, act, 1);
  };
  // gcn_rnn concat (shared by z and r): cat198 = [x, h1, h2] with dual adjacency
  auto gcn_rnn_cat = [&](const float* combp) {
    const int C = F + H;       // 66
    const int catLd = 3 * C;   // 198
    ew3(combp, (long)N * C, C, nullptr, nullptr, 1.f, 0.f, 0.f, cat198, catLd, 0, C);
    gconv(adpbuf, (long)N * N, cat198, catLd, C, t1buf);
    gconv(adj, 0, cat198, catLd, C, t2buf);
    ew3(combp, (long)N * C, C, t1buf, t2buf, AL_C, BE_C, GA_C, cat198, catLd, C, C);
    gconv(adpbuf, (long)N * N, cat198 + C, catLd, C, t1buf);
    gconv(adj, 0, cat198 + C, catLd, C, t2buf);
    ew3(combp, (long)N * C, C, t1buf, t2buf, AL_C, BE_C, GA_C, cat198, catLd, 2 * C, C);
  };

  auto gru_step = [&](const float* cur, long curBS) {
    // gate terms gs/gt from hidden state with static adjacency
    gcn_single(hidden, (long)N * H, H, H, adj, 0, cat192, agg1_w, agg1_b, HYP, gsb, 0, 0);
    gcn_single(hidden, (long)N * H, H, H, adj, 0, cat192, agg2_w, agg2_b, HYP, gtb, 0, 0);
    {
      long total = rows * HYP;
      int blk = (int)((total + 255) / 256);
      hipLaunchKernelGGL(gate_tanh_kernel, dim3(blk), dim3(256), 0, stream,
                         cur, curBS, src_w, src_b, gsb, nsb, N, rows);
      hipLaunchKernelGGL(gate_tanh_kernel, dim3(blk), dim3(256), 0, stream,
                         cur, curBS, tgt_w, tgt_b, gtb, ntb, N, rows);
    }
    // adaptive adjacency: a = ns@nt^T - nt@ns^T ; then relu(tanh)+eye, normalize
    gemm(nsb, ntb, nullptr, adpbuf, N, N, HYP, (long)N * HYP, (long)N * HYP, (long)N * N,
         HYP, HYP, N, 0, 1, 0, 1.f, 0, 0, B);
    gemm(ntb, nsb, nullptr, adpbuf, N, N, HYP, (long)N * HYP, (long)N * HYP, (long)N * N,
         HYP, HYP, N, 0, 1, 0, -1.f, 1, 0, B);
    hipLaunchKernelGGL(adp_norm_kernel, dim3(N, B), dim3(32), 0, stream, adpbuf, N);
    // comb = [cur, hidden]
    ew3(cur, curBS, F, nullptr, nullptr, 1.f, 0.f, 0.f, comb, 66, 0, F);
    ew3(hidden, (long)N * H, H, nullptr, nullptr, 1.f, 0.f, 0.f, comb, 66, F, H);
    gcn_rnn_cat(comb);
    gemm(cat198, gz_w, gz_b, zb, (int)rows, H, 198, 0, 0, 0, 198, H, H, 0, 0, 0, 1.f, 0, 2, 1);
    gemm(cat198, gr_w, gr_b, rb, (int)rows, H, 198, 0, 0, 0, 198, H, H, 0, 0, 0, 1.f, 0, 2, 1);
    // comb2 = [cur, r*hidden]  (slice0 already holds cur)
    {
      long total = rows * H;
      int blk = (int)((total + 255) / 256);
      hipLaunchKernelGGL(mul_slice_kernel, dim3(blk), dim3(256), 0, stream,
                         rb, hidden, comb, 66, F, H, rows);
    }
    gcn_rnn_cat(comb);
    gemm(cat198, gc_w, gc_b, cb, (int)rows, H, 198, 0, 0, 0, 198, H, H, 0, 0, 0, 1.f, 0, 3, 1);
    {
      long total = rows * H;
      int blk = (int)((total + 255) / 256);
      hipLaunchKernelGGL(gru_update_kernel, dim3(blk), dim3(256), 0, stream,
                         hidden, zb, cb, total);
    }
  };

  // ---- forward pass ----
  hipMemsetAsync(hidden, 0, (size_t)B * N * H * sizeof(float), stream);

  for (int t = 0; t < T - 4; t += 4)
    gru_step(X + (long)t * N * F, (long)T * N * F);

  // attention q/k over the last 4 frames
  {
    long tq = (long)B * N * D;
    hipLaunchKernelGGL(qk_lin_kernel, dim3((int)((tq + 255) / 256)), dim3(256), 0, stream,
                       X, wq_w, wq_b, qb, B, 23, 1, N, D, T);
    long tk = (long)B * 4 * N * D;
    hipLaunchKernelGGL(qk_lin_kernel, dim3((int)((tk + 255) / 256)), dim3(256), 0, stream,
                       X, wk_w, wk_b, kb, B, 20, 4, N, D, T);
  }

  // TGN: per source frame build A_tgn = preA + attn, graph-conv, relu, sum into tar2
  for (int tt = 0; tt < 4; ++tt) {
    hipLaunchKernelGGL(attn_kernel, dim3(N, B), dim3(32), 0, stream,
                       qb, kb, trans_w, adj, adpbuf, N, tt);
    const float* srcT = X + (long)(20 + tt) * N * F;
    gcn_single(srcT, (long)T * N * F, F, F, adpbuf, (long)N * N, cat6,
               tgn_w, tgn_b, TGN, tar2, /*relu*/1, /*accum*/ tt > 0 ? 1 : 0);
  }

  // gat = tar2 @ fcs_w + fcs_b
  gemm(tar2, fcs_w, fcs_b, gatb, (int)rows, F, TGN, 0, 0, 0, TGN, F, F, 0, 0, 0, 1.f, 0, 0, 1);

  // final GRU step with gat as input
  gru_step(gatb, (long)N * F);

  // out = ([tar2, hidden] @ fcm_w + fcm_b).transpose(0,2,1) -> (B, 24, N)
  ew3(tar2, (long)N * TGN, TGN, nullptr, nullptr, 1.f, 0.f, 0.f, cat96, 96, 0, TGN);
  ew3(hidden, (long)N * H, H, nullptr, nullptr, 1.f, 0.f, 0.f, cat96, 96, TGN, H);
  gemm(cat96, fcm_w, fcm_b, out, N, LEN, 96, (long)N * 96, 0, (long)LEN * N,
       96, LEN, N, 0, 0, /*transOut*/1, 1.f, 0, 0, B);
}
